// LSTMCharTagger_23785528885812
// MI455X (gfx1250) — matrix operand order repaired and checked by reading, not compiled, not run on hardware
//
#include <hip/hip_runtime.h>
#include <hip/hip_bf16.h>

// ---------------------------------------------------------------------------
// LSTMCharTagger for MI455X (gfx1250, wave32, WMMA).
//   S=2048 words, L=16 chars, Dc=64, Hc=256, Dw=256, Hw=512, T=64
// Phases:
//   1) pack weights into WMMA B-fragment tile layout (bf16)
//   2) char LSTM: batched over 2048 words, 16 steps, v_wmma_f32_16x16x32_bf16
//      (weight fragment loads kept inside the timestep loop via an opaque
//       per-iteration pointer offset -> regular L2-cacheable global_load_b128,
//       no LICM hoist/scratch spill, no volatile SCOPE_SYS serialization)
//   3) word_in A-fragment pack; G_x = word_in @ Ww_ih^T + bw  (WMMA GEMM)
//   4) sequential word LSTM (batch=1): persistent block, VALU matvec vs L2
//   5) tag projection + log_softmax (f32)
// ---------------------------------------------------------------------------

typedef __attribute__((ext_vector_type(16))) __bf16        v16bf;
typedef __attribute__((ext_vector_type(8)))  float         v8f;
typedef __attribute__((ext_vector_type(4)))  unsigned int  u32x4;

#define S_LEN   2048
#define L_CH    16
#define D_CHAR  64
#define H_CHAR  256
#define D_WORD  256
#define H_WORD  512
#define T_TAGS  64

// ---- workspace layout (bytes, all 256-aligned) ----
#define OFF_WCIH   ((size_t)0)          // 2*64   tiles * 512 bf16 = 131072 B
#define OFF_WCHH   ((size_t)131072)     // 8*64   tiles            = 524288 B
#define OFF_WWIH   ((size_t)655360)     // 16*128 tiles            = 2097152 B
#define OFF_WWHHT  ((size_t)2752512)    // 512*2048 f32            = 4194304 B
#define OFF_HCHAR  ((size_t)6946816)    // 2048*256 bf16           = 1048576 B
#define OFF_APACK  ((size_t)7995392)    // 128*16 tiles            = 2097152 B
#define OFF_GX     ((size_t)10092544)   // 2048*2048 f32           = 16777216 B
#define OFF_HS     ((size_t)26869760)   // 2048*512 f32            = 4194304 B

// ---- helpers ----
__device__ __forceinline__ unsigned short f2bf(float f) {
    unsigned int u = __float_as_uint(f);
    unsigned int r = u + 0x7FFFu + ((u >> 16) & 1u);   // round-to-nearest-even
    return (unsigned short)(r >> 16);
}
__device__ __forceinline__ float bf2f(unsigned short s) {
    return __uint_as_float(((unsigned int)s) << 16);
}
__device__ __forceinline__ float sigm(float x) { return 1.0f / (1.0f + expf(-x)); }
__device__ __forceinline__ v8f splat8(float v) {
    v8f r;
#pragma unroll
    for (int i = 0; i < 8; ++i) r[i] = v;
    return r;
}

// A-fragment K index for 16-bit A 16x32 (ISA 7.12.2): lanes0-15 K∈{0..7,16..23},
// lanes16-31 K∈{8..15,24..31}
__device__ __forceinline__ int amap(int v, int p, int hi) {
    return (v < 4) ? (8 * hi + 2 * v + p) : (16 + 8 * hi + 2 * (v - 4) + p);
}

// gather an A fragment from LDS bf16 [rows][rowstride], tile base k0
// (two contiguous 8-ushort runs -> compiler emits 2x ds_load_b128)
__device__ __forceinline__ v16bf load_a_lds(const unsigned short* base, int row,
                                            int rowstride, int k0, int hi) {
    union { unsigned short s[16]; v16bf v; } u;
#pragma unroll
    for (int vv = 0; vv < 8; ++vv)
#pragma unroll
        for (int p = 0; p < 2; ++p)
            u.s[2 * vv + p] = base[row * rowstride + k0 + amap(vv, p, hi)];
    return u.v;
}

// load a packed fragment (512 bf16/tile, lane-major 16 bf16 contiguous) -> 2x b128
__device__ __forceinline__ v16bf load_pack(const unsigned short* pack, int tile, int lane) {
    union { u32x4 q[2]; v16bf v; } u;
    const u32x4* p = (const u32x4*)(pack + (size_t)tile * 512 + lane * 16);
    u.q[0] = p[0];
    u.q[1] = p[1];
    return u.v;
}

// ---------------------------------------------------------------------------
// Pack W (row-major [N,K] f32) as W^T B-fragment tiles: tile = kt*ntiles+nt,
// within tile: [lane][slot], B map (ISA 7.12.2): N=nt*16+lane%16, K=kt*32+amap(v,p,hi)
// ---------------------------------------------------------------------------
__global__ void pack_b_kernel(const float* __restrict__ W, unsigned short* __restrict__ dst,
                              int Ktot, int ktiles, int ntiles) {
    int total = ktiles * ntiles * 512;
    for (int idx = blockIdx.x * blockDim.x + threadIdx.x; idx < total;
         idx += gridDim.x * blockDim.x) {
        int tile = idx >> 9, rem = idx & 511;
        int lane = rem >> 4, slot = rem & 15;
        int kt = tile / ntiles, nt = tile % ntiles;
        int l16 = lane & 15, hi = lane >> 4;
        int vv = slot >> 1, p = slot & 1;
        int k = kt * 32 + amap(vv, p, hi);
        int n = nt * 16 + l16;
        dst[idx] = f2bf(W[(size_t)n * Ktot + k]);
    }
}

// Ww_hh [2048,512] -> k-major f32 [512][2048] for coalesced recurrent matvec
__global__ void transpose_wwhh_kernel(const float* __restrict__ W, float* __restrict__ dst) {
    int total = 2048 * 512;
    for (int idx = blockIdx.x * blockDim.x + threadIdx.x; idx < total;
         idx += gridDim.x * blockDim.x) {
        int n = idx / 512, k = idx - n * 512;
        dst[(size_t)k * 2048 + n] = W[idx];
    }
}

// ---------------------------------------------------------------------------
// Char LSTM: 128 workgroups x 256 threads (8 waves). WG owns 16 words.
// Wave wv owns hidden units j∈[32wv,32wv+32) for all four gates:
//   8 gate tiles + 2 cell tiles + 2 hidden tiles live in VGPRs (D-layout).
// ---------------------------------------------------------------------------
__global__ __launch_bounds__(256) void char_lstm_kernel(
    const int* __restrict__ char_ixs, const int* __restrict__ char_lengths,
    const float* __restrict__ char_emb,
    const unsigned short* __restrict__ wcih_pack,
    const unsigned short* __restrict__ wchh_pack,
    const float* __restrict__ bc_ih, const float* __restrict__ bc_hh,
    unsigned short* __restrict__ hchar_bf) {
    __shared__ unsigned short h_bf[16][H_CHAR];
    __shared__ unsigned short x_bf[16][D_CHAR];
    __shared__ int lens[16];

    const int tid = threadIdx.x;
    const int wv = tid >> 5, lane = tid & 31;
    const int l16 = lane & 15, hi = lane >> 4;
    const int w0 = blockIdx.x * 16;

    if (tid < 16) lens[tid] = char_lengths[w0 + tid];
    for (int e = tid; e < 16 * H_CHAR; e += 256) (&h_bf[0][0])[e] = 0;

    float bias[8];
#pragma unroll
    for (int q = 0; q < 8; ++q) {
        int gate = q >> 1, tt = q & 1;
        int col = gate * H_CHAR + 32 * wv + 16 * tt + l16;
        bias[q] = bc_ih[col] + bc_hh[col];
    }

    v8f cacc[2], hacc[2];
    cacc[0] = splat8(0.0f);
    cacc[1] = splat8(0.0f);
    hacc[0] = splat8(0.0f);
    hacc[1] = splat8(0.0f);

    for (int t = 0; t < L_CH; ++t) {
        // opaque zero offset: compiler must assume it changes per iteration,
        // so weight-fragment loads stay inside the loop (no hoist -> no spill)
        // while remaining plain L2-cacheable global_load_b128.
        unsigned opq = 0;
        asm volatile("" : "+s"(opq));
        const unsigned short* wih = wcih_pack + opq;
        const unsigned short* whh = wchh_pack + opq;

        __syncthreads();  // prev h writes visible; x_bf free to rewrite
        // stage x_t = char_emb[char_ixs[w][t]] as bf16
        for (int e = tid; e < 16 * D_CHAR; e += 256) {
            int m = e >> 6, d = e & 63;
            int ci = char_ixs[(w0 + m) * L_CH + t];
            (&x_bf[0][0])[e] = f2bf(char_emb[(size_t)ci * D_CHAR + d]);
        }
        __syncthreads();

        v8f acc[8];
#pragma unroll
        for (int q = 0; q < 8; ++q) acc[q] = splat8(bias[q]);

        // input contribution: K = 64 (2 k-steps)
#pragma unroll
        for (int kt = 0; kt < 2; ++kt) {
            v16bf a = load_a_lds(&x_bf[0][0], l16, D_CHAR, kt * 32, hi);
#pragma unroll
            for (int q = 0; q < 8; ++q) {
                int gate = q >> 1, tt = q & 1;
                int nt = gate * 16 + 2 * wv + tt;  // (gate*256+32wv+16tt)/16
                v16bf b = load_pack(wih, kt * 64 + nt, lane);
                acc[q] = __builtin_amdgcn_wmma_f32_16x16x32_bf16(
                    false, a, false, b, (short)0, acc[q], false, false);
            }
        }
        // hidden contribution: K = 256 (8 k-steps)
#pragma unroll
        for (int kt = 0; kt < 8; ++kt) {
            v16bf a = load_a_lds(&h_bf[0][0], l16, H_CHAR, kt * 32, hi);
#pragma unroll
            for (int q = 0; q < 8; ++q) {
                int gate = q >> 1, tt = q & 1;
                int nt = gate * 16 + 2 * wv + tt;
                v16bf b = load_pack(whh, kt * 64 + nt, lane);
                acc[q] = __builtin_amdgcn_wmma_f32_16x16x32_bf16(
                    false, a, false, b, (short)0, acc[q], false, false);
            }
        }
        __syncthreads();  // all h_bf reads done before rewrite

        // elementwise LSTM update (wave-local i/f/g/o, c and h), ragged mask
#pragma unroll
        for (int tt = 0; tt < 2; ++tt) {
#pragma unroll
            for (int vv = 0; vv < 8; ++vv) {
                int m = vv + 8 * hi;             // word row (D-layout)
                int j = 32 * wv + 16 * tt + l16; // hidden unit
                float iv = sigm(acc[0 + tt][vv]);
                float fv = sigm(acc[2 + tt][vv]);
                float gv = tanhf(acc[4 + tt][vv]);
                float ov = sigm(acc[6 + tt][vv]);
                float c_old = cacc[tt][vv];
                float h_old = hacc[tt][vv];
                float c_new = fv * c_old + iv * gv;
                float h_new = ov * tanhf(c_new);
                bool keep = (t < lens[m]);
                float c_sel = keep ? c_new : c_old;
                float h_sel = keep ? h_new : h_old;
                cacc[tt][vv] = c_sel;
                hacc[tt][vv] = h_sel;
                h_bf[m][j] = f2bf(h_sel);
            }
        }
    }
    __syncthreads();
    for (int e = tid; e < 16 * H_CHAR; e += 256) {
        int m = e >> 8, j = e & 255;
        hchar_bf[(size_t)(w0 + m) * H_CHAR + j] = h_bf[m][j];
    }
}

// ---------------------------------------------------------------------------
// Pack word_in = [word_emb | h_char] as WMMA A-fragment tiles (tile = mt*16+kt)
// ---------------------------------------------------------------------------
__global__ void pack_a_wordin_kernel(const int* __restrict__ word_ixs,
                                     const float* __restrict__ word_emb,
                                     const unsigned short* __restrict__ hchar_bf,
                                     unsigned short* __restrict__ dst) {
    int total = 128 * 16 * 512;
    for (int idx = blockIdx.x * blockDim.x + threadIdx.x; idx < total;
         idx += gridDim.x * blockDim.x) {
        int tile = idx >> 9, rem = idx & 511;
        int lane = rem >> 4, slot = rem & 15;
        int mt = tile >> 4, kt = tile & 15;
        int l16 = lane & 15, hi = lane >> 4;
        int vv = slot >> 1, p = slot & 1;
        int k = kt * 32 + amap(vv, p, hi);
        int m = mt * 16 + l16;
        unsigned short val;
        if (k < D_WORD)
            val = f2bf(word_emb[(size_t)word_ixs[m] * D_WORD + k]);
        else
            val = hchar_bf[(size_t)m * H_CHAR + (k - D_WORD)];
        dst[idx] = val;
    }
}

// ---------------------------------------------------------------------------
// G_x = word_in @ Ww_ih^T + (bw_ih + bw_hh): M=2048, N=2048, K=512. WMMA.
// grid = (128 mtiles, 16 ntile-groups), wave wv -> ntile = 8*group + wv.
// ---------------------------------------------------------------------------
__global__ __launch_bounds__(256) void word_gemm_x_kernel(
    const unsigned short* __restrict__ a_pack, const unsigned short* __restrict__ b_pack,
    const float* __restrict__ bw_ih, const float* __restrict__ bw_hh,
    float* __restrict__ gx) {
    const int tid = threadIdx.x;
    const int wv = tid >> 5, lane = tid & 31;
    const int l16 = lane & 15, hi = lane >> 4;
    const int mt = blockIdx.x;
    const int nt = blockIdx.y * 8 + wv;
    const int n = nt * 16 + l16;

    v8f acc = splat8(bw_ih[n] + bw_hh[n]);
#pragma unroll
    for (int kt = 0; kt < 16; ++kt) {
        v16bf a = load_pack(a_pack, mt * 16 + kt, lane);
        v16bf b = load_pack(b_pack, kt * 128 + nt, lane);
        acc = __builtin_amdgcn_wmma_f32_16x16x32_bf16(false, a, false, b, (short)0,
                                                      acc, false, false);
    }
#pragma unroll
    for (int vv = 0; vv < 8; ++vv) {
        int m = mt * 16 + vv + 8 * hi;
        gx[(size_t)m * 2048 + n] = acc[vv];
    }
}

// ---------------------------------------------------------------------------
// Sequential word LSTM (batch=1): one persistent 1024-thread block.
// gates[t] = G_x[t] + h @ Wwhh^T; thread n owns gate cols n and n+1024.
// ---------------------------------------------------------------------------
__global__ __launch_bounds__(1024) void word_recurrent_kernel(
    const float* __restrict__ gx, const float* __restrict__ wwhh_t,
    float* __restrict__ hs) {
    __shared__ float h_s[H_WORD];
    __shared__ float f_s[H_WORD];
    __shared__ float o_s[H_WORD];
    const int tid = threadIdx.x;
    if (tid < H_WORD) h_s[tid] = 0.0f;
    float c_reg = 0.0f;
    __syncthreads();

    for (int t = 0; t < S_LEN; ++t) {
        const float* grow = gx + (size_t)t * 2048;
        if (t + 1 < S_LEN) {
            __builtin_prefetch(gx + (size_t)(t + 1) * 2048 + tid, 0, 1);
        }
        float acc0 = grow[tid];
        float acc1 = grow[tid + 1024];
#pragma unroll 8
        for (int k = 0; k < H_WORD; ++k) {
            float hk = h_s[k];
            const float* wr = wwhh_t + (size_t)k * 2048;
            acc0 = fmaf(hk, wr[tid], acc0);
            acc1 = fmaf(hk, wr[tid + 1024], acc1);
        }
        __syncthreads();  // all h_s reads done
        if (tid >= H_WORD) {  // forget & output gates
            f_s[tid - H_WORD] = sigm(acc0);
            o_s[tid - H_WORD] = sigm(acc1);
        }
        __syncthreads();
        if (tid < H_WORD) {  // input & cell gates + state update
            float iv = sigm(acc0);
            float gv = tanhf(acc1);
            float c_new = f_s[tid] * c_reg + iv * gv;
            float h_new = o_s[tid] * tanhf(c_new);
            c_reg = c_new;
            h_s[tid] = h_new;
            hs[(size_t)t * H_WORD + tid] = h_new;
        }
        __syncthreads();
    }
}

// ---------------------------------------------------------------------------
// tag_space = hs @ W_tag^T + b_tag; log_softmax over T=64. 1 word / block.
// ---------------------------------------------------------------------------
__global__ __launch_bounds__(64) void tagger_kernel(const float* __restrict__ hs,
                                                    const float* __restrict__ Wtag,
                                                    const float* __restrict__ btag,
                                                    float* __restrict__ out) {
    __shared__ float red[T_TAGS];
    const int t = blockIdx.x;
    const int j = threadIdx.x;
    const float* h = hs + (size_t)t * H_WORD;
    float acc = btag[j];
    for (int k = 0; k < H_WORD; ++k) acc = fmaf(h[k], Wtag[j * H_WORD + k], acc);

    red[j] = acc;
    __syncthreads();
    for (int s = 32; s > 0; s >>= 1) {
        if (j < s) red[j] = fmaxf(red[j], red[j + s]);
        __syncthreads();
    }
    float m = red[0];
    __syncthreads();
    red[j] = expf(acc - m);
    __syncthreads();
    for (int s = 32; s > 0; s >>= 1) {
        if (j < s) red[j] += red[j + s];
        __syncthreads();
    }
    float lse = m + logf(red[0]);
    out[(size_t)t * T_TAGS + j] = acc - lse;
}

// ---------------------------------------------------------------------------
extern "C" void kernel_launch(void* const* d_in, const int* in_sizes, int n_in,
                              void* d_out, int out_size, void* d_ws, size_t ws_size,
                              hipStream_t stream) {
    (void)in_sizes; (void)n_in; (void)out_size; (void)ws_size;
    const int*   word_ixs     = (const int*)d_in[0];
    const int*   char_ixs     = (const int*)d_in[1];
    const int*   char_lengths = (const int*)d_in[2];
    const float* word_emb     = (const float*)d_in[3];
    const float* char_emb     = (const float*)d_in[4];
    const float* Wc_ih        = (const float*)d_in[5];
    const float* Wc_hh        = (const float*)d_in[6];
    const float* bc_ih        = (const float*)d_in[7];
    const float* bc_hh        = (const float*)d_in[8];
    const float* Ww_ih        = (const float*)d_in[9];
    const float* Ww_hh        = (const float*)d_in[10];
    const float* bw_ih        = (const float*)d_in[11];
    const float* bw_hh        = (const float*)d_in[12];
    const float* W_tag        = (const float*)d_in[13];
    const float* b_tag        = (const float*)d_in[14];
    float* out = (float*)d_out;
    char*  ws  = (char*)d_ws;

    unsigned short* wcih_pack = (unsigned short*)(ws + OFF_WCIH);
    unsigned short* wchh_pack = (unsigned short*)(ws + OFF_WCHH);
    unsigned short* wwih_pack = (unsigned short*)(ws + OFF_WWIH);
    float*          wwhh_t    = (float*)(ws + OFF_WWHHT);
    unsigned short* hchar_bf  = (unsigned short*)(ws + OFF_HCHAR);
    unsigned short* a_pack    = (unsigned short*)(ws + OFF_APACK);
    float*          gx        = (float*)(ws + OFF_GX);
    float*          hs        = (float*)(ws + OFF_HS);

    pack_b_kernel<<<128, 256, 0, stream>>>(Wc_ih, wcih_pack, D_CHAR, 2, 64);
    pack_b_kernel<<<256, 256, 0, stream>>>(Wc_hh, wchh_pack, H_CHAR, 8, 64);
    pack_b_kernel<<<512, 256, 0, stream>>>(Ww_ih, wwih_pack, 512, 16, 128);
    transpose_wwhh_kernel<<<512, 256, 0, stream>>>(Ww_hh, wwhh_t);

    char_lstm_kernel<<<128, 256, 0, stream>>>(char_ixs, char_lengths, char_emb,
                                              wcih_pack, wchh_pack, bc_ih, bc_hh,
                                              hchar_bf);

    pack_a_wordin_kernel<<<512, 256, 0, stream>>>(word_ixs, word_emb, hchar_bf, a_pack);

    word_gemm_x_kernel<<<dim3(128, 16), 256, 0, stream>>>(a_pack, wwih_pack,
                                                          bw_ih, bw_hh, gx);

    word_recurrent_kernel<<<1, 1024, 0, stream>>>(gx, wwhh_t, hs);

    tagger_kernel<<<S_LEN, T_TAGS, 0, stream>>>(hs, W_tag, b_tag, out);
}
